// LocalAttention_34488587387515
// MI455X (gfx1250) — matrix operand order repaired and checked by reading, not compiled, not run on hardware
//
#include <hip/hip_runtime.h>

// Local (banded) attention, flash-attention style, CDNA5 wave32 WMMA.
// B=2 H=16 L=2048 D=64, window +/-128, temperature 8.
// Softmax in log2 domain (fused 1/8*log2e scale, native v_exp_f32);
// row reductions via DPP (no ds_bpermute, keeps LDS port free for P/V tiles).

#define LSEQ   2048
#define DHEAD  64
#define WIN    128
#define QTILE  128
#define NCHUNK 12            // keys [qbase-128, qbase+255] in 32-key chunks
#define SCL2   0.18033688011112042f   // (1/8) * log2(e)
#define NEGBIG (-1e9f)
#define PSTR   40            // LDS row stride in halves (80B: 16B-aligned, conflict-free)

typedef __attribute__((ext_vector_type(16))) _Float16 v16h;
typedef __attribute__((ext_vector_type(8)))  _Float16 v8h;
typedef __attribute__((ext_vector_type(8)))  float    v8f;

#define WMMA(A, B, C) \
  __builtin_amdgcn_wmma_f32_16x16x32_f16(false, (A), false, (B), (short)0, (C), false, false)

__device__ __forceinline__ v16h pack16(float4 a, float4 b, float4 c, float4 d) {
  v16h r;
  r[0]  = (_Float16)a.x; r[1]  = (_Float16)a.y; r[2]  = (_Float16)a.z; r[3]  = (_Float16)a.w;
  r[4]  = (_Float16)b.x; r[5]  = (_Float16)b.y; r[6]  = (_Float16)b.z; r[7]  = (_Float16)b.w;
  r[8]  = (_Float16)c.x; r[9]  = (_Float16)c.y; r[10] = (_Float16)c.z; r[11] = (_Float16)c.w;
  r[12] = (_Float16)d.x; r[13] = (_Float16)d.y; r[14] = (_Float16)d.z; r[15] = (_Float16)d.w;
  return r;
}

__device__ __forceinline__ v16h cat8(v8h lo, v8h hi) {
  return __builtin_shufflevector(lo, hi, 0,1,2,3,4,5,6,7,8,9,10,11,12,13,14,15);
}

// A-matrix 16x32 f16 layout (ISA 7.12.2): lane m=l%16, hi=l/16,
// element e -> K = 8*hi + e + (e<8 ? 0 : 8).  Two contiguous 8-float runs.
__device__ __forceinline__ v16h loadA(const float* row, int hi, int dbase) {
  const float4* p = (const float4*)(row + dbase);
  return pack16(p[2*hi], p[2*hi + 1], p[4 + 2*hi], p[5 + 2*hi]);
}

// B-matrix 32x16 f16 layout: lane n=l%16, hi=l/16,
// element e -> K = 8*hi + e + (e<8 ? 0 : 8).  Two contiguous 8-float runs.
__device__ __forceinline__ v16h loadB(const float* row, int hi, int dbase) {
  const float4* p = (const float4*)(row + dbase);
  return pack16(p[2*hi], p[2*hi + 1], p[4 + 2*hi], p[5 + 2*hi]);
}

// DPP lane-swap within a 16-lane row (our reduction groups are exactly DPP rows).
template <int CTRL>
__device__ __forceinline__ float dpp_swap(float x) {
  return __int_as_float(
      __builtin_amdgcn_update_dpp(0, __float_as_int(x), CTRL, 0xf, 0xf, true));
}
// Butterfly over 16 lanes: xor1, xor2, then (after quads are uniform)
// row_half_mirror (xor7 == xor4) and row_mirror (xor15 == xor8).
__device__ __forceinline__ float rmax16(float x) {
  x = fmaxf(x, dpp_swap<0x0B1>(x));   // quad_perm [1,0,3,2]
  x = fmaxf(x, dpp_swap<0x04E>(x));   // quad_perm [2,3,0,1]
  x = fmaxf(x, dpp_swap<0x141>(x));   // row_half_mirror
  x = fmaxf(x, dpp_swap<0x140>(x));   // row_mirror
  return x;
}
__device__ __forceinline__ float rsum16(float x) {
  x += dpp_swap<0x0B1>(x);
  x += dpp_swap<0x04E>(x);
  x += dpp_swap<0x141>(x);
  x += dpp_swap<0x140>(x);
  return x;
}

__global__ __launch_bounds__(256) void local_attn_wmma(const float* __restrict__ q,
                                                       const float* __restrict__ k,
                                                       const float* __restrict__ v,
                                                       float* __restrict__ out) {
  // Vt: V chunk transposed+converted, [d][key]; Pbuf: per-wave P tile [m][key].
  __shared__ __align__(16) _Float16 Vt[DHEAD * PSTR];
  __shared__ __align__(16) _Float16 Pbuf[8][16 * PSTR];

  const int tid   = threadIdx.x;
  const int wave  = tid >> 5;
  const int lane  = tid & 31;
  const int hi    = lane >> 4;       // half-wave select
  const int ln    = lane & 15;       // column/row within 16
  const int bh    = blockIdx.y;
  const int qbase = blockIdx.x * QTILE;
  const int wq0   = qbase + wave * 16;   // this wave's first query row

  const float* qp = q + (size_t)bh * LSEQ * DHEAD;
  const float* kp = k + (size_t)bh * LSEQ * DHEAD;
  const float* vp = v + (size_t)bh * LSEQ * DHEAD;
  float*       op = out + (size_t)bh * LSEQ * DHEAD;

  _Float16* Pw = &Pbuf[wave][0];

  // Q fragments for this wave's 16 rows (A layout), d = 0..31 and 32..63.
  const float* qr  = qp + (size_t)(wq0 + ln) * DHEAD;
  const v16h qf0 = loadA(qr, hi, 0);
  const v16h qf1 = loadA(qr, hi, 32);

  v8f   O[4] = {v8f{}, v8f{}, v8f{}, v8f{}};
  float row_max[8], row_sum[8];    // log2-domain running max / sum
#pragma unroll
  for (int r = 0; r < 8; ++r) { row_max[r] = NEGBIG; row_sum[r] = 0.0f; }

  for (int c = 0; c < NCHUNK; ++c) {
    const int kchunk = qbase - WIN + c * 32;

    // ---- cooperative stage: V chunk -> LDS, f32->f16, transposed ----
#pragma unroll
    for (int it = 0; it < 2; ++it) {
      int idx = tid + it * 256;          // 512 float4s = 32 keys x 64 d
      int row = idx >> 4;                // key within chunk
      int c4  = idx & 15;                // float4 index along d
      int gk  = kchunk + row;
      gk = gk < 0 ? 0 : (gk >= LSEQ ? LSEQ - 1 : gk);
      float4 f = ((const float4*)(vp + (size_t)gk * DHEAD))[c4];
      int d0 = c4 * 4;
      Vt[(d0 + 0) * PSTR + row] = (_Float16)f.x;
      Vt[(d0 + 1) * PSTR + row] = (_Float16)f.y;
      Vt[(d0 + 2) * PSTR + row] = (_Float16)f.z;
      Vt[(d0 + 3) * PSTR + row] = (_Float16)f.w;
    }
    __syncthreads();

    // Prefetch next chunk's K/V rows into GL2 while we compute.
    if (c + 1 < NCHUNK) {
      int nk = kchunk + 32 + lane;
      nk = nk < 0 ? 0 : (nk >= LSEQ ? LSEQ - 1 : nk);
      __builtin_prefetch(kp + (size_t)nk * DHEAD, 0, 3);
      __builtin_prefetch(vp + (size_t)nk * DHEAD, 0, 3);
    }

    // Wave-uniform skip: this 32-key chunk intersects this wave's band iff
    // kchunk+31 >= wq0-WIN and kchunk <= wq0+15+WIN.  EXEC stays all-1s inside.
    const bool active = (kchunk + 31 >= wq0 - WIN) && (kchunk <= wq0 + 15 + WIN);
    if (active) {
      // ---- S = Q K^T for two 16-key subtiles ----
      v8f accS[2];
#pragma unroll
      for (int sub = 0; sub < 2; ++sub) {
        int krow = kchunk + sub * 16 + ln;
        krow = krow < 0 ? 0 : (krow >= LSEQ ? LSEQ - 1 : krow);
        const float* kr = kp + (size_t)krow * DHEAD;
        v16h kf0 = loadB(kr, hi, 0);
        v16h kf1 = loadB(kr, hi, 32);
        v8f acc = {};
        acc = WMMA(qf0, kf0, acc);
        acc = WMMA(qf1, kf1, acc);
        accS[sub] = acc;
      }

      // ---- banded mask + online softmax in log2 domain ----
      float alpha[8];
#pragma unroll
      for (int r = 0; r < 8; ++r) {
        const int gm = wq0 + 8 * hi + r;
        const int n0 = kchunk + ln;
        const int n1 = kchunk + 16 + ln;
        int d0 = gm - n0; d0 = d0 < 0 ? -d0 : d0;
        int d1 = gm - n1; d1 = d1 < 0 ? -d1 : d1;
        const bool ok0 = (n0 >= 0) && (n0 < LSEQ) && (d0 <= WIN);
        const bool ok1 = (n1 >= 0) && (n1 < LSEQ) && (d1 <= WIN);
        float s0 = ok0 ? accS[0][r] * SCL2 : NEGBIG;   // log2-units
        float s1 = ok1 ? accS[1][r] * SCL2 : NEGBIG;

        float mx   = rmax16(fmaxf(s0, s1));
        float mnew = fmaxf(row_max[r], mx);
        float al   = __builtin_amdgcn_exp2f(row_max[r] - mnew);
        float p0   = __builtin_amdgcn_exp2f(s0 - mnew);
        float p1   = __builtin_amdgcn_exp2f(s1 - mnew);
        row_sum[r] = row_sum[r] * al + rsum16(p0 + p1);
        row_max[r] = mnew;
        alpha[r]   = al;

        Pw[(8 * hi + r) * PSTR + ln]      = (_Float16)p0;
        Pw[(8 * hi + r) * PSTR + 16 + ln] = (_Float16)p1;
      }
#pragma unroll
      for (int t = 0; t < 4; ++t)
#pragma unroll
        for (int r = 0; r < 8; ++r) O[t][r] *= alpha[r];

      // Fence per-wave P write -> read (same-wave LDS, ordered via DScnt).
      asm volatile("s_wait_dscnt 0x0" ::: "memory");

      // ---- O += P V : A = P (16x32, A layout from LDS), B = V^T tiles ----
      v8h plo = *(const v8h*)(Pw + ln * PSTR + 8 * hi);
      v8h phi = *(const v8h*)(Pw + ln * PSTR + 16 + 8 * hi);
      v16h pf = cat8(plo, phi);
#pragma unroll
      for (int t = 0; t < 4; ++t) {
        const _Float16* vb = Vt + (t * 16 + ln) * PSTR + 8 * hi;
        v16h vf = cat8(*(const v8h*)vb, *(const v8h*)(vb + 16));
        O[t] = WMMA(pf, vf, O[t]);
      }
    }
    __syncthreads();   // protect Vt before next chunk's staging
  }

  // ---- normalize and store ----
#pragma unroll
  for (int r = 0; r < 8; ++r) {
    const float inv = 1.0f / row_sum[r];
    const int   gm  = wq0 + 8 * hi + r;
    float* orow = op + (size_t)gm * DHEAD;
#pragma unroll
    for (int t = 0; t < 4; ++t) orow[t * 16 + ln] = O[t][r] * inv;
  }
}

extern "C" void kernel_launch(void* const* d_in, const int* in_sizes, int n_in,
                              void* d_out, int out_size, void* d_ws, size_t ws_size,
                              hipStream_t stream) {
  const float* q = (const float*)d_in[0];
  const float* k = (const float*)d_in[1];
  const float* v = (const float*)d_in[2];
  float* out = (float*)d_out;
  const int BH = in_sizes[0] / (LSEQ * DHEAD);   // B*H = 32
  dim3 grid(LSEQ / QTILE, BH);
  local_attn_wmma<<<grid, 256, 0, stream>>>(q, k, v, out);
}
